// LinearAttention_75385265979625
// MI455X (gfx1250) — matrix operand order repaired and checked by reading, not compiled, not run on hardware
//
#include <hip/hip_runtime.h>

// Problem constants
#define BB 256
#define NN 200
#define CC 512
#define HH 8
#define DD 64
#define NP 224          // N padded to multiple of 32
#define QKC 1024        // 2*C

typedef __attribute__((ext_vector_type(16))) _Float16 v16h;
typedef __attribute__((ext_vector_type(8)))  _Float16 v8h;
typedef __attribute__((ext_vector_type(8)))  float    v8f;

__device__ __forceinline__ v16h cat16(v8h lo, v8h hi) {
    return __builtin_shufflevector(lo, hi, 0,1,2,3,4,5,6,7,8,9,10,11,12,13,14,15);
}

// WMMA f32 += f16 * f16, 16x16x32
__device__ __forceinline__ v8f wmma_f16(v16h a, v16h b, v8f c) {
    return __builtin_amdgcn_wmma_f32_16x16x32_f16(false, a, false, b, (short)0, c, false, false);
}

// Async copy of 16 bytes global -> LDS (CDNA5 GLOBAL_LOAD_ASYNC_TO_LDS_B128,
// tracked by ASYNCcnt).  ldsoff = LDS byte offset (low 32 bits of flat shared ptr).
__device__ __forceinline__ void async_g2l_b128(uint32_t ldsoff, const void* gptr) {
    asm volatile("global_load_async_to_lds_b128 %0, %1, off"
                 :: "v"(ldsoff), "v"(gptr) : "memory");
}
__device__ __forceinline__ void wait_async0() {
    asm volatile("s_wait_asynccnt 0x0" ::: "memory");
}

// A-fragment (16x32, MxK) from row-major matrix (pitch in halves).
// Lane m = lane&15 holds row m; lane<16: K = {k0..k0+15},
// lane>=16: K = {k0+16..k0+31}.  All chunks 16B aligned.
__device__ __forceinline__ v16h load_a_rm(const _Float16* base, int row0, int pitch,
                                          int k0, int lane) {
    int m  = lane & 15;
    int kb = (lane < 16) ? 0 : 16;
    const _Float16* p = base + (size_t)(row0 + m) * pitch + k0 + kb;
    v8h lo = *(const v8h*)(p);
    v8h hi = *(const v8h*)(p + 8);
    return cat16(lo, hi);
}

// B-fragment (32x16, KxN) where memory holds B^T row-major (per-column contiguous K).
__device__ __forceinline__ v16h load_b_colK(const _Float16* base, int col0, int pitch,
                                            int k0, int lane) {
    int n  = lane & 15;
    int kb = (lane < 16) ? 0 : 16;
    const _Float16* p = base + (size_t)(col0 + n) * pitch + k0 + kb;
    return *(const v16h*)p; // 32B contiguous
}

// B-fragment (32x16, KxN) from row-major B (strided per-lane gather).
__device__ __forceinline__ v16h load_b_rm(const _Float16* base, int k0, int col0,
                                          int pitch, int lane) {
    int n  = lane & 15;
    int kb = (lane < 16) ? 0 : 16;
    const _Float16* p = base + (size_t)(k0 + kb) * pitch + col0 + n;
    v16h r;
#pragma unroll
    for (int j = 0; j < 16; ++j) r[j] = p[(size_t)j * pitch];
    return r;
}

// ---------------- K1: conversions / init ----------------
__global__ void k_convert_x(const float* __restrict__ x, _Float16* __restrict__ xh) {
    size_t i = (size_t)blockIdx.x * 256 + threadIdx.x;
    if (i >= (size_t)BB * NP * CC) return;
    int c = (int)(i % CC);
    size_t t = i / CC;
    int n = (int)(t % NP);
    size_t b = t / NP;
    float v = (n < NN) ? x[(b * NN + n) * CC + c] : 0.f;
    xh[i] = (_Float16)v;
}

__global__ void k_convert_w(const float* __restrict__ Wqk, _Float16* __restrict__ wht) {
    size_t i = (size_t)blockIdx.x * 256 + threadIdx.x;
    if (i >= (size_t)QKC * CC) return;
    int col = (int)(i / CC);
    int k   = (int)(i % CC);
    wht[i] = (_Float16)Wqk[(size_t)k * QKC + col];
}

__global__ void k_zero_f32(float* __restrict__ p, int n) {
    int i = blockIdx.x * 256 + threadIdx.x;
    if (i < n) p[i] = 0.f;
}

// ---------------- K2: qk GEMM + bias + elu+1 + RoPE + kmean accum ----------------
// grid (B, 7), block 256 (8 waves).  Block: rows [nt*32, nt*32+32), cols 0..1023.
// Wave w owns col tiles [8w, 8w+8), processed as RoPE pairs (t, t+2).
__global__ __launch_bounds__(256) void k_gemm_qk(
    const _Float16* __restrict__ xh, const _Float16* __restrict__ wht,
    const float* __restrict__ bqk,
    _Float16* __restrict__ q_r, _Float16* __restrict__ krt,
    float* __restrict__ ksum)
{
    __shared__ _Float16 smem[32 * 520];
    const int b  = blockIdx.x;
    const int n0 = blockIdx.y * 32;
    const int tid = threadIdx.x, wave = tid >> 5, lane = tid & 31;

    // Async staging of the 32x512 f16 x-tile into LDS (16B chunks, ASYNCcnt path)
    const _Float16* src = xh + ((size_t)b * NP + n0) * CC;
#pragma unroll
    for (int it = 0; it < 8; ++it) {
        int chunk = tid + it * 256;          // 2048 chunks of 8 halves
        int r = chunk >> 6;
        int off = (chunk & 63) * 8;
        uint32_t ldsoff = (uint32_t)(uintptr_t)(smem + r * 520 + off);
        async_g2l_b128(ldsoff, src + (size_t)r * CC + off);
    }
    wait_async0();
    __syncthreads();

    const int pairA[4] = {0, 1, 4, 5};
    for (int pg = 0; pg < 4; ++pg) {
        const int tA = wave * 8 + pairA[pg];
        const int cA = tA * 16;                 // first tile col; (cA%64)<32 guaranteed
        const _Float16* wcol0 = wht + (size_t)(cA + (lane & 15)) * CC;
        v8f acc[2][2] = {};                     // [row-tile][pair member]
        for (int k0 = 0; k0 < CC; k0 += 32) {
            if (k0 + 32 < CC) {                 // prefetch next K-chunk of W (near cache)
                __builtin_prefetch(wcol0 + k0 + 32, 0, 3);
                __builtin_prefetch(wcol0 + 32 * CC + k0 + 32, 0, 3);
            }
            v16h bf0 = load_b_colK(wht, cA,      CC, k0, lane);
            v16h bf1 = load_b_colK(wht, cA + 32, CC, k0, lane);
#pragma unroll
            for (int rt = 0; rt < 2; ++rt) {
                v16h af = load_a_rm(smem, rt * 16, 520, k0, lane);
                acc[rt][0] = wmma_f16(af, bf0, acc[rt][0]);
                acc[rt][1] = wmma_f16(af, bf1, acc[rt][1]);
            }
        }
        // Epilogue
        const int colq = cA + (lane & 15);       // global col of member 0
        const bool isq = (colq < CC);
        const int c512 = colq & (CC - 1);
        const int h  = c512 >> 6;
        const int dd = c512 & (DD - 1);          // < 32 by construction
        const float invfreq = __expf(-0.28782313662425572f * (float)dd); // 10000^(-dd/32)
        const float bias0 = bqk[colq], bias1 = bqk[colq + 32];
        const int mbase = (lane < 16) ? 0 : 8;
        float sum0 = 0.f, sum1 = 0.f;
#pragma unroll
        for (int rt = 0; rt < 2; ++rt) {
#pragma unroll
            for (int i = 0; i < 8; ++i) {
                int n = n0 + rt * 16 + mbase + i;
                float v0 = acc[rt][0][i] + bias0;
                float v1 = acc[rt][1][i] + bias1;
                float e0 = (v0 > 0.f) ? v0 + 1.f : __expf(v0); // elu(x)+1
                float e1 = (v1 > 0.f) ? v1 + 1.f : __expf(v1);
                float s, co;
                __sincosf((float)n * invfreq, &s, &co);
                float r0 = e0 * co - e1 * s;
                float r1 = e1 * co + e0 * s;
                bool valid = (n < NN);
                if (isq) {
                    _Float16* dst = q_r + ((size_t)b * NP + n) * CC + c512;
                    dst[0]  = valid ? (_Float16)r0 : (_Float16)0.f;
                    dst[32] = valid ? (_Float16)r1 : (_Float16)0.f;
                } else {
                    _Float16* dst = krt + ((size_t)(b * HH + h) * DD + dd) * NP + n;
                    dst[0]            = valid ? (_Float16)(r0 * (1.f / NN)) : (_Float16)0.f;
                    dst[32 * NP]      = valid ? (_Float16)(r1 * (1.f / NN)) : (_Float16)0.f;
                    if (valid) { sum0 += e0; sum1 += e1; }
                }
            }
        }
        if (!isq) {
            float* ks = ksum + (size_t)(b * HH + h) * DD + dd;
            atomicAdd(ks, sum0);
            atomicAdd(ks + 32, sum1);
        }
    }
}

// ---------------- K3: z = 1/(<q,kmean>+eps) via rotation invariance ----------------
// grid B*N, block 256: wave h handles head h, lane = dd (0..31).
__global__ __launch_bounds__(256) void k_z(
    const _Float16* __restrict__ q_r, const float* __restrict__ ksum,
    float* __restrict__ z)
{
    int bn = blockIdx.x;
    int b = bn / NN, n = bn % NN;
    int h = threadIdx.x >> 5, dd = threadIdx.x & 31;
    const _Float16* q = q_r + ((size_t)b * NP + n) * CC + h * DD;
    float q0 = (float)q[dd], q1 = (float)q[dd + 32];
    const float* ks = ksum + (size_t)(b * HH + h) * DD;
    float km0 = ks[dd] * (1.f / NN), km1 = ks[dd + 32] * (1.f / NN);
    float invfreq = __expf(-0.28782313662425572f * (float)dd);
    float s, co;
    __sincosf((float)n * invfreq, &s, &co);
    // <q_unroped, km> == <q_r, R_n km>
    float part = q0 * (km0 * co - km1 * s) + q1 * (km1 * co + km0 * s);
#pragma unroll
    for (int off = 16; off; off >>= 1) part += __shfl_xor(part, off, 32);
    if (dd == 0) z[(size_t)(b * HH + h) * NN + n] = 1.f / (part + 1e-6f);
}

// ---------------- K4: kv[b,h] = krt(64x224) @ v(224x64) ----------------
// grid B*H, block 256 (8 waves), wave owns 2 of 16 output 16x16 tiles.
__global__ __launch_bounds__(256) void k_kv(
    const _Float16* __restrict__ krt, const _Float16* __restrict__ xh,
    _Float16* __restrict__ kv)
{
    int bh = blockIdx.x;
    int b = bh >> 3, h = bh & 7;
    int wave = threadIdx.x >> 5, lane = threadIdx.x & 31;
    const _Float16* A = krt + (size_t)bh * DD * NP;           // 64 x 224 row-major
    const _Float16* V = xh + (size_t)b * NP * CC + h * DD;    // 224 x 64, pitch CC
#pragma unroll
    for (int ti = 0; ti < 2; ++ti) {
        int t = wave * 2 + ti;
        int m0 = (t >> 2) * 16, e0 = (t & 3) * 16;
        v8f acc = {};
        for (int k0 = 0; k0 < NP; k0 += 32) {
            v16h af = load_a_rm(A, m0, NP, k0, lane);
            v16h bf = load_b_rm(V, k0, e0, CC, lane);
            acc = wmma_f16(af, bf, acc);
        }
        int col = e0 + (lane & 15);
        int mb = m0 + ((lane < 16) ? 0 : 8);
        _Float16* dst = kv + (size_t)bh * DD * DD + (size_t)mb * DD + col;
#pragma unroll
        for (int i = 0; i < 8; ++i) dst[(size_t)i * DD] = (_Float16)acc[i];
    }
}

// ---------------- K5: out = (q_r @ kv) * z + lepe(x) ----------------
// grid (B, 13), block 256 (8 waves). Block: rows [nt*16, nt*16+16), cols 0..511.
__global__ __launch_bounds__(256) void k_out(
    const _Float16* __restrict__ q_r, const _Float16* __restrict__ kv,
    const float* __restrict__ z, const float* __restrict__ x,
    const float* __restrict__ lepe_w, const float* __restrict__ lepe_b,
    float* __restrict__ out)
{
    int b = blockIdx.x;
    int n0 = blockIdx.y * 16;
    int wave = threadIdx.x >> 5, lane = threadIdx.x & 31;
#pragma unroll
    for (int ti = 0; ti < 4; ++ti) {
        int t = wave * 4 + ti;               // 32 col tiles
        int col0 = t * 16;
        int h = col0 >> 6;
        int e0 = col0 & (DD - 1);
        const _Float16* A  = q_r + ((size_t)b * NP + n0) * CC + h * DD; // 16x64, pitch CC
        const _Float16* Bm = kv + (size_t)(b * HH + h) * DD * DD;       // 64x64 row-major
        v8f acc = {};
        for (int k0 = 0; k0 < DD; k0 += 32) {
            v16h af = load_a_rm(A, 0, CC, k0, lane);
            v16h bf = load_b_rm(Bm, k0, e0, DD, lane);
            acc = wmma_f16(af, bf, acc);
        }
        int col = col0 + (lane & 15);
        float w0 = lepe_w[col * 3 + 0], w1 = lepe_w[col * 3 + 1], w2 = lepe_w[col * 3 + 2];
        float lb = lepe_b[col];
        int mbase = (lane < 16) ? 0 : 8;
#pragma unroll
        for (int i = 0; i < 8; ++i) {
            int n = n0 + mbase + i;
            if (n >= NN) continue;
            float zz = z[(size_t)(b * HH + h) * NN + n];
            const float* xp = x + ((size_t)b * NN + n) * CC + col;
            float xm1 = (n > 0)      ? xp[-(int)CC] : 0.f;
            float xc  = xp[0];
            float xp1 = (n < NN - 1) ? xp[CC]       : 0.f;
            out[((size_t)b * NN + n) * CC + col] = acc[i] * zz + w0 * xm1 + w1 * xc + w2 * xp1 + lb;
        }
    }
}

// ---------------- host launch ----------------
extern "C" void kernel_launch(void* const* d_in, const int* in_sizes, int n_in,
                              void* d_out, int out_size, void* d_ws, size_t ws_size,
                              hipStream_t stream) {
    const float* x      = (const float*)d_in[0];
    const float* Wqk    = (const float*)d_in[1];
    const float* bqk    = (const float*)d_in[2];
    const float* lepe_w = (const float*)d_in[3];
    const float* lepe_b = (const float*)d_in[4];
    float* out = (float*)d_out;

    char* ws = (char*)d_ws;
    size_t off = 0;
    auto take = [&](size_t bytes) { size_t o = off; off = (off + bytes + 255) & ~(size_t)255; return o; };
    _Float16* xh   = (_Float16*)(ws + take((size_t)BB * NP * CC * 2));
    _Float16* wht  = (_Float16*)(ws + take((size_t)QKC * CC * 2));
    _Float16* q_r  = (_Float16*)(ws + take((size_t)BB * NP * CC * 2));
    _Float16* krt  = (_Float16*)(ws + take((size_t)BB * HH * DD * NP * 2));
    float*    ksum = (float*)   (ws + take((size_t)BB * HH * DD * 4));
    float*    zbuf = (float*)   (ws + take((size_t)BB * HH * NN * 4));
    _Float16* kv   = (_Float16*)(ws + take((size_t)BB * HH * DD * DD * 2));
    (void)ws_size; (void)in_sizes; (void)n_in; (void)out_size;

    {
        size_t n = (size_t)BB * NP * CC;
        k_convert_x<<<dim3((unsigned)((n + 255) / 256)), 256, 0, stream>>>(x, xh);
    }
    {
        size_t n = (size_t)QKC * CC;
        k_convert_w<<<dim3((unsigned)((n + 255) / 256)), 256, 0, stream>>>(Wqk, wht);
    }
    k_zero_f32<<<dim3((BB * HH * DD + 255) / 256), 256, 0, stream>>>(ksum, BB * HH * DD);

    k_gemm_qk<<<dim3(BB, NP / 32), 256, 0, stream>>>(xh, wht, bqk, q_r, krt, ksum);
    k_z<<<dim3(BB * NN), 256, 0, stream>>>(q_r, ksum, zbuf);
    k_kv<<<dim3(BB * HH), 256, 0, stream>>>(krt, xh, kv);
    k_out<<<dim3(BB, 13), 256, 0, stream>>>(q_r, kv, zbuf, x, lepe_w, lepe_b, out);
}
